// BasicModel_56160992362860
// MI455X (gfx1250) — compile-verified, matrix operand order checked by reading, not run on hardware
//
#include <hip/hip_runtime.h>
#include <hip/hip_bf16.h>
#include <math.h>

// ---------------------------------------------------------------------------
// LSTM caption decoder for MI455X (gfx1250, wave32, WMMA).
// bf16 WMMA 16x16x32 with f32 accumulation; each wave computes a 64x32 tile
// (4Mx2N register tiling -> 8 WMMAs per 12 b128 loads instead of 8 per 32).
// ---------------------------------------------------------------------------

typedef __attribute__((ext_vector_type(16))) __bf16 v16bf;
typedef __attribute__((ext_vector_type(8)))  float  v8f;
typedef __bf16 bf16_t;

#define Bx 256
#define MAXLEN 31
#define Tn 30
#define Vn 10000
#define VnPad 10240        // 640 x 16 : N padded so 32-wide N tiles divide evenly
#define Hn 512
#define En 512
#define KCAT 1536          // E (image) + E (embedding) + H (hidden)
#define NGATE 2048         // 4*H

// ---------------- WMMA fragment load (ISA 7.12.2 16-bit A/B layout) ----------
// For lane l: half = l>>4 selects K sub-block; 16 bf16 elements are two
// contiguous 8-element (16-byte) runs: [k0 + half*8 .. +7] and [k0+16+half*8 .. +7].
__device__ __forceinline__ v16bf load_frag16(const bf16_t* rowk0, int half) {
    union { v16bf v; uint4 q[2]; } u;
    u.q[0] = *reinterpret_cast<const uint4*>(rowk0 + half * 8);
    u.q[1] = *reinterpret_cast<const uint4*>(rowk0 + 16 + half * 8);
    return u.v;
}

// ---------------- one-time setup kernels ------------------------------------
__global__ void k_sort_meta(const int* __restrict__ lens,
                            const int* __restrict__ w_input,
                            int* __restrict__ order,
                            int* __restrict__ dec_len,
                            float* __restrict__ out_target,
                            float* __restrict__ out_declen) {
    __shared__ int ord[Bx];
    const int b = threadIdx.x;
    const int L = lens[b];
    int rank = 0;
    for (int j = 0; j < Bx; ++j) {
        const int Lj = lens[j];
        rank += (Lj > L) || (Lj == L && j < b);   // stable descending
    }
    ord[rank] = b;
    __syncthreads();
    const int src = ord[b];
    order[b] = src;
    const int dl = lens[src] - 1;
    dec_len[b] = dl;
    out_declen[b] = (float)dl;
    const int* wrow = w_input + src * MAXLEN;
    for (int j = 0; j < Tn; ++j) out_target[b * Tn + j] = (float)wrow[j + 1];
}

__global__ void k_build_wcat(const float* __restrict__ W_ih,
                             const float* __restrict__ W_hh,
                             bf16_t* __restrict__ Wcat) {
    const int idx = blockIdx.x * blockDim.x + threadIdx.x;   // 2048*1536
    const int j = idx / KCAT, k = idx - j * KCAT;
    const float v = (k < 2 * En) ? W_ih[j * (2 * En) + k]
                                 : W_hh[j * Hn + (k - 2 * En)];
    Wcat[idx] = (bf16_t)v;
}

__global__ void k_build_wout(const float* __restrict__ W_out,
                             bf16_t* __restrict__ Wo) {
    const int idx = blockIdx.x * blockDim.x + threadIdx.x;   // 10240*512
    const int n = idx / Hn;
    Wo[idx] = (n < Vn) ? (bf16_t)W_out[idx] : (bf16_t)0.0f;  // zero pad rows
}

__global__ void k_build_bias(const float* __restrict__ b_ih,
                             const float* __restrict__ b_hh,
                             float* __restrict__ bias) {
    const int i = blockIdx.x * blockDim.x + threadIdx.x;     // 2048
    bias[i] = b_ih[i] + b_hh[i];
}

__global__ void k_init_state(const float* __restrict__ g_img,
                             const int* __restrict__ order,
                             bf16_t* __restrict__ A_step,
                             float* __restrict__ h,
                             float* __restrict__ c) {
    const int idx = blockIdx.x * blockDim.x + threadIdx.x;   // 256*512
    const int b = idx >> 9, k = idx & 511;
    A_step[b * KCAT + k]            = (bf16_t)g_img[order[b] * En + k]; // image part
    A_step[b * KCAT + 2 * En + k]   = (bf16_t)0.0f;                     // h part (h0=0)
    h[idx] = 0.0f;
    c[idx] = 0.0f;
}

__global__ void k_zero_pad(float* __restrict__ preds) {      // preds[:, T, :] = 0
    const int idx = blockIdx.x * blockDim.x + threadIdx.x;   // 256*10000
    const int b = idx / Vn, v = idx - b * Vn;
    preds[((size_t)b * MAXLEN + Tn) * Vn + v] = 0.0f;
}

// ---------------- per-step kernels ------------------------------------------
__global__ void k_prep_emb(const float* __restrict__ emb,
                           const int* __restrict__ w_input,
                           const int* __restrict__ order,
                           bf16_t* __restrict__ A_step, int t) {
    const int idx = blockIdx.x * blockDim.x + threadIdx.x;   // 256*512
    const int b = idx >> 9, k = idx & 511;
    const int word = w_input[order[b] * MAXLEN + t];
    A_step[b * KCAT + En + k] = (bf16_t)emb[word * En + k];
}

// --------- 64x32 register-tiled WMMA GEMM body (4 M-tiles x 2 N-tiles) ------
// C[64x32] += A[64xK] * B(=W^T)[Kx32], A/W row-major bf16 with leading dim K.
template <int K, int LDC>
__device__ __forceinline__ void wmma_tile_64x32(const bf16_t* __restrict__ A,
                                                const bf16_t* __restrict__ W,
                                                int tM, int tN, int lane,
                                                v8f (&acc)[8]) {
    const int half = lane >> 4, r = lane & 15;
    const bf16_t* arow0 = A + (tM +  0 + r) * K;
    const bf16_t* arow1 = A + (tM + 16 + r) * K;
    const bf16_t* arow2 = A + (tM + 32 + r) * K;
    const bf16_t* arow3 = A + (tM + 48 + r) * K;
    const bf16_t* brow0 = W + (tN +  0 + r) * K;   // B[k][n] = W[n][k]
    const bf16_t* brow1 = W + (tN + 16 + r) * K;
    for (int k0 = 0; k0 < K; k0 += 32) {
        v16bf a0 = load_frag16(arow0 + k0, half);
        v16bf a1 = load_frag16(arow1 + k0, half);
        v16bf a2 = load_frag16(arow2 + k0, half);
        v16bf a3 = load_frag16(arow3 + k0, half);
        v16bf b0 = load_frag16(brow0 + k0, half);
        v16bf b1 = load_frag16(brow1 + k0, half);
        acc[0] = __builtin_amdgcn_wmma_f32_16x16x32_bf16(false, a0, false, b0, (short)0, acc[0], false, false);
        acc[1] = __builtin_amdgcn_wmma_f32_16x16x32_bf16(false, a1, false, b0, (short)0, acc[1], false, false);
        acc[2] = __builtin_amdgcn_wmma_f32_16x16x32_bf16(false, a2, false, b0, (short)0, acc[2], false, false);
        acc[3] = __builtin_amdgcn_wmma_f32_16x16x32_bf16(false, a3, false, b0, (short)0, acc[3], false, false);
        acc[4] = __builtin_amdgcn_wmma_f32_16x16x32_bf16(false, a0, false, b1, (short)0, acc[4], false, false);
        acc[5] = __builtin_amdgcn_wmma_f32_16x16x32_bf16(false, a1, false, b1, (short)0, acc[5], false, false);
        acc[6] = __builtin_amdgcn_wmma_f32_16x16x32_bf16(false, a2, false, b1, (short)0, acc[6], false, false);
        acc[7] = __builtin_amdgcn_wmma_f32_16x16x32_bf16(false, a3, false, b1, (short)0, acc[7], false, false);
    }
}

// gates = A_step(256x1536,bf16) @ Wcat^T(1536x2048) + bias ; f32 accumulate
__global__ void k_gemm_gates(const bf16_t* __restrict__ A,
                             const bf16_t* __restrict__ W,
                             const float* __restrict__ bias,
                             float* __restrict__ gates) {
    const int lane = threadIdx.x & 31;
    const int wave = threadIdx.x >> 5;
    const int tile = blockIdx.x * 8 + wave;       // 4 Mgrp * 64 Ngrp = 256 tiles
    const int tN = (tile & 63) * 32;              // N = 2048 -> 64 groups of 32
    const int tM = (tile >> 6) * 64;              // M = 256  -> 4 groups of 64
    v8f acc[8] = {};
    wmma_tile_64x32<KCAT, NGATE>(A, W, tM, tN, lane, acc);
    const int half = lane >> 4, r = lane & 15;
#pragma unroll
    for (int j = 0; j < 2; ++j) {
        const int n = tN + 16 * j + r;
        const float bv = bias[n];
#pragma unroll
        for (int q = 0; q < 4; ++q) {
#pragma unroll
            for (int i = 0; i < 8; ++i) {          // D: m = i + 8*half
                gates[(tM + 16 * q + i + 8 * half) * NGATE + n] = acc[4 * j + q][i] + bv;
            }
        }
    }
}

__device__ __forceinline__ float sigmoidf_(float x) { return 1.0f / (1.0f + expf(-x)); }

__global__ void k_lstm_cell(const float* __restrict__ gates,
                            float* __restrict__ h, float* __restrict__ c,
                            bf16_t* __restrict__ hnew_bf,
                            bf16_t* __restrict__ A_step,
                            const int* __restrict__ dec_len, int t) {
    const int idx = blockIdx.x * blockDim.x + threadIdx.x;   // 256*512
    const int b = idx >> 9, k = idx & 511;
    const float* gr = gates + b * NGATE;
    const float ig = sigmoidf_(gr[k]);
    const float fg = sigmoidf_(gr[Hn + k]);
    const float gg = tanhf(gr[2 * Hn + k]);
    const float og = sigmoidf_(gr[3 * Hn + k]);
    const float cprev = c[idx];
    const float cn = fg * cprev + ig * gg;
    const float hn = og * tanhf(cn);
    hnew_bf[idx] = (bf16_t)hn;                    // logits always use h_new
    const bool active = t < dec_len[b];
    const float hv = active ? hn : h[idx];
    const float cv = active ? cn : cprev;
    h[idx] = hv;
    c[idx] = cv;
    A_step[b * KCAT + 2 * En + k] = (bf16_t)hv;   // masked h feeds next step
}

// logits = h_new(256x512,bf16) @ W_out^T(512x10240pad) + b_out -> preds[:, t, :]
__global__ void k_gemm_out(const bf16_t* __restrict__ A,
                           const bf16_t* __restrict__ W,
                           const float* __restrict__ b_out,
                           float* __restrict__ preds, int t) {
    const int lane = threadIdx.x & 31;
    const int wave = threadIdx.x >> 5;
    const int tile = blockIdx.x * 8 + wave;       // 4 Mgrp * 320 Ngrp = 1280 tiles
    const int tN = (tile % 320) * 32;             // Npad = 10240 -> 320 groups of 32
    const int tM = (tile / 320) * 64;             // M = 256 -> 4 groups of 64
    v8f acc[8] = {};
    wmma_tile_64x32<Hn, Vn>(A, W, tM, tN, lane, acc);
    const int half = lane >> 4, r = lane & 15;
#pragma unroll
    for (int j = 0; j < 2; ++j) {
        const int n = tN + 16 * j + r;
        if (n >= Vn) continue;                     // padded region: skip store
        const float bv = b_out[n];
#pragma unroll
        for (int q = 0; q < 4; ++q) {
#pragma unroll
            for (int i = 0; i < 8; ++i) {
                const int m = tM + 16 * q + i + 8 * half;
                preds[((size_t)m * MAXLEN + t) * Vn + n] = acc[4 * j + q][i] + bv;
            }
        }
    }
}

__global__ void k_log_softmax(float* __restrict__ preds,
                              const int* __restrict__ dec_len, int t) {
    const int b = blockIdx.x;
    float* row = preds + ((size_t)b * MAXLEN + t) * Vn;
    const bool active = t < dec_len[b];
    __shared__ float red[256];
    if (!active) {
        for (int v = threadIdx.x; v < Vn; v += 256) row[v] = 0.0f;
        return;
    }
    float m = -3.402823466e38f;
    for (int v = threadIdx.x; v < Vn; v += 256) m = fmaxf(m, row[v]);
    red[threadIdx.x] = m;
    __syncthreads();
    for (int s = 128; s > 0; s >>= 1) {
        if (threadIdx.x < s) red[threadIdx.x] = fmaxf(red[threadIdx.x], red[threadIdx.x + s]);
        __syncthreads();
    }
    m = red[0];
    __syncthreads();
    float sum = 0.0f;
    for (int v = threadIdx.x; v < Vn; v += 256) sum += expf(row[v] - m);
    red[threadIdx.x] = sum;
    __syncthreads();
    for (int s = 128; s > 0; s >>= 1) {
        if (threadIdx.x < s) red[threadIdx.x] += red[threadIdx.x + s];
        __syncthreads();
    }
    const float lse = m + logf(red[0]);
    for (int v = threadIdx.x; v < Vn; v += 256) row[v] = row[v] - lse;
}

// ---------------- host-side orchestration -----------------------------------
static inline size_t align256(size_t x) { return (x + 255) & ~(size_t)255; }

extern "C" void kernel_launch(void* const* d_in, const int* in_sizes, int n_in,
                              void* d_out, int out_size, void* d_ws, size_t ws_size,
                              hipStream_t stream) {
    const float* g_img  = (const float*)d_in[0];   // (256, 512)
    const int*   w_inp  = (const int*)  d_in[1];   // (256, 31)
    const int*   lens   = (const int*)  d_in[2];   // (256,)
    const float* emb    = (const float*)d_in[3];   // (10000, 512)
    const float* W_ih   = (const float*)d_in[4];   // (2048, 1024)
    const float* W_hh   = (const float*)d_in[5];   // (2048, 512)
    const float* b_ih   = (const float*)d_in[6];   // (2048,)
    const float* b_hh   = (const float*)d_in[7];   // (2048,)
    const float* W_out  = (const float*)d_in[8];   // (10000, 512)
    const float* b_out  = (const float*)d_in[9];   // (10000,)

    float* preds      = (float*)d_out;                                  // 256*31*10000
    float* out_target = preds + (size_t)Bx * MAXLEN * Vn;               // 256*30
    float* out_declen = out_target + (size_t)Bx * Tn;                   // 256

    // workspace carve-up (~21 MB)
    char* ws = (char*)d_ws;
    size_t off = 0;
    bf16_t* Wcat   = (bf16_t*)(ws + off); off += align256((size_t)NGATE * KCAT * 2);
    bf16_t* Wo     = (bf16_t*)(ws + off); off += align256((size_t)VnPad * Hn * 2);
    bf16_t* A_step = (bf16_t*)(ws + off); off += align256((size_t)Bx * KCAT * 2);
    bf16_t* hnewbf = (bf16_t*)(ws + off); off += align256((size_t)Bx * Hn * 2);
    float*  h      = (float*) (ws + off); off += align256((size_t)Bx * Hn * 4);
    float*  c      = (float*) (ws + off); off += align256((size_t)Bx * Hn * 4);
    float*  gates  = (float*) (ws + off); off += align256((size_t)Bx * NGATE * 4);
    float*  bias   = (float*) (ws + off); off += align256((size_t)NGATE * 4);
    int*    order  = (int*)   (ws + off); off += align256((size_t)Bx * 4);
    int*    declen = (int*)   (ws + off); off += align256((size_t)Bx * 4);

    // setup
    k_sort_meta <<<1, Bx, 0, stream>>>(lens, w_inp, order, declen, out_target, out_declen);
    k_build_wcat<<<(NGATE * KCAT) / 256, 256, 0, stream>>>(W_ih, W_hh, Wcat);
    k_build_wout<<<(VnPad * Hn) / 256, 256, 0, stream>>>(W_out, Wo);
    k_build_bias<<<NGATE / 256, 256, 0, stream>>>(b_ih, b_hh, bias);
    k_init_state<<<(Bx * Hn) / 256, 256, 0, stream>>>(g_img, order, A_step, h, c);
    k_zero_pad  <<<(Bx * Vn) / 256, 256, 0, stream>>>(preds);

    // recurrence
    for (int t = 0; t < Tn; ++t) {
        k_prep_emb   <<<(Bx * Hn) / 256, 256, 0, stream>>>(emb, w_inp, order, A_step, t);
        k_gemm_gates <<<(4 * 64) / 8, 256, 0, stream>>>(A_step, Wcat, bias, gates);
        k_lstm_cell  <<<(Bx * Hn) / 256, 256, 0, stream>>>(gates, h, c, hnewbf, A_step, declen, t);
        k_gemm_out   <<<(4 * 320) / 8, 256, 0, stream>>>(hnewbf, Wo, b_out, preds, t);
        k_log_softmax<<<Bx, 256, 0, stream>>>(preds, declen, t);
    }
}
